// AttentionBiasHead_35175782154703
// MI455X (gfx1250) — matrix operand order repaired and checked by reading, not compiled, not run on hardware
//
#include <hip/hip_runtime.h>
#include <hip/hip_bf16.h>
#include <math.h>

// Problem dims
#define B_    32
#define L_    512
#define DIN   512
#define DQ    128
#define DS_   256
#define DMLP  128
#define LL    (L_*L_)   // 262144

typedef __attribute__((ext_vector_type(16))) __bf16 v16bf;
typedef __attribute__((ext_vector_type(8)))  __bf16 v8bf;
typedef __attribute__((ext_vector_type(8)))  float  v8f;

union V16U { v16bf v; v8bf h[2]; };

// Native bf16 converts (RNE) -> hardware cvt instead of bit-twiddle VALU.
__device__ __forceinline__ unsigned short f2bfu(float f) {
  return __builtin_bit_cast(unsigned short, static_cast<__bf16>(f));
}

// ---------------------------------------------------------------------------
// Pack W [K, N] fp32 -> W^T [N, K] bf16 (K contiguous -> clean WMMA B loads)
// ---------------------------------------------------------------------------
__global__ __launch_bounds__(256) void pack_wT(const float* __restrict__ W,
                                               unsigned short* __restrict__ WT,
                                               int K, int N) {
  int i = blockIdx.x * 256 + threadIdx.x;
  if (i >= K * N) return;
  int k = i / N, n = i % N;
  WT[(size_t)n * K + k] = f2bfu(W[i]);
}

// ---------------------------------------------------------------------------
// Projections: Y = X @ W + b   (M=B*L=16384, N=128, K=512)
// Block stages its 16x512 fp32 A tile once into LDS as bf16 (cooperative,
// coalesced), then 8 waves (one 16-col N tile each) run 16 WMMAs from LDS.
// mode 0: q -> row-major bf16, scaled by 1/sqrt(128)
// mode 1: k -> row-major bf16
// mode 2: v -> transposed bf16 vT[b][d][l]  (for attn@v B operand)
// ---------------------------------------------------------------------------
__global__ __launch_bounds__(256) void qkv_gemm(const float* __restrict__ X,
                                                const unsigned short* __restrict__ WT,
                                                const float* __restrict__ bias,
                                                unsigned short* __restrict__ Yrm,
                                                unsigned short* __restrict__ Yt,
                                                int mode) {
  __shared__ __align__(32) unsigned short xs[16][DIN + 8];  // 16x520 u16, row pitch 1040B (16B mult)

  const int tid = threadIdx.x;
  const int m0 = blockIdx.x * 16;

  // Cooperative load+convert: 16 rows x 512 fp32 -> bf16 in LDS. float4 per op.
  for (int i = tid; i < 16 * (DIN / 4); i += 256) {
    int r = i >> 7;          // DIN/4 = 128 float4 per row
    int c = (i & 127) * 4;
    float4 f = *(const float4*)(X + (size_t)(m0 + r) * DIN + c);
    xs[r][c + 0] = f2bfu(f.x);
    xs[r][c + 1] = f2bfu(f.y);
    xs[r][c + 2] = f2bfu(f.z);
    xs[r][c + 3] = f2bfu(f.w);
  }
  __syncthreads();

  const int lane = tid & 31;
  const int wave = tid >> 5;
  const int n0 = wave * 16;
  const int lm = lane & 15;
  const int kg = lane >> 4;

  const unsigned short* wrow = WT + (size_t)(n0 + lm) * DIN;

  v8f acc = {};
  for (int k0 = 0; k0 < DIN; k0 += 32) {
    V16U a;
    a.h[0] = *(const v8bf*)&xs[lm][k0 + kg * 16];
    a.h[1] = *(const v8bf*)&xs[lm][k0 + 8 + kg * 16];
    v16bf b = *(const v16bf*)(wrow + k0 + 16 * kg);
    acc = __builtin_amdgcn_wmma_f32_16x16x32_bf16(false, a.v, false, b,
                                                  (short)0, acc, false, false);
  }

  const float bn = bias[n0 + lm];
  const float qscale = 0.08838834764831845f;  // 1/sqrt(128)
#pragma unroll
  for (int e = 0; e < 8; e++) {
    float vv = acc[e] + bn;
    if (mode == 0) vv *= qscale;
    int m = m0 + e + 8 * kg;
    if (mode == 2) {
      int bb = m >> 9, l = m & (L_ - 1);
      Yt[(size_t)bb * DQ * L_ + (size_t)(n0 + lm) * L_ + l] = f2bfu(vv);
    } else {
      Yrm[(size_t)m * DQ + n0 + lm] = f2bfu(vv);
    }
  }
}

// ---------------------------------------------------------------------------
// h = relu(sf @ Wb1 + bb1)   [32,256]@[256,128] -> [32,128]  (tiny, fp32 VALU)
// ---------------------------------------------------------------------------
__global__ __launch_bounds__(256) void mlp1(const float* __restrict__ sf,
                                            const float* __restrict__ Wb1,
                                            const float* __restrict__ bb1,
                                            float* __restrict__ h) {
  int i = blockIdx.x * 256 + threadIdx.x;
  if (i >= B_ * DMLP) return;
  int b = i / DMLP, j = i % DMLP;
  float acc = bb1[j];
  const float* s = sf + (size_t)b * DS_;
  for (int k = 0; k < DS_; k++) acc = fmaf(s[k], Wb1[(size_t)k * DMLP + j], acc);
  h[i] = fmaxf(acc, 0.0f);
}

// ---------------------------------------------------------------------------
// bias = h @ Wb2 + bb2  (M=32, K=128, N=262144): HBM-bound on Wb2 (134 MB).
// Thread owns one n, streams Wb2 rows coalesced, 32 batch accumulators.
// ---------------------------------------------------------------------------
__global__ __launch_bounds__(256) void bias2(const float* __restrict__ h,
                                             const float* __restrict__ Wb2,
                                             const float* __restrict__ bb2,
                                             float* __restrict__ biasOut) {
  __shared__ float hs[B_ * DMLP];  // 16 KB
  for (int i = threadIdx.x; i < B_ * DMLP; i += 256) hs[i] = h[i];
  __syncthreads();

  const int n = blockIdx.x * 256 + threadIdx.x;
  float acc[B_];
#pragma unroll
  for (int b = 0; b < B_; b++) acc[b] = 0.0f;

  for (int k = 0; k < DMLP; k++) {
    float w = Wb2[(size_t)k * LL + n];
    if (k + 1 < DMLP) __builtin_prefetch(Wb2 + (size_t)(k + 1) * LL + n, 0, 1);
#pragma unroll
    for (int b = 0; b < B_; b++) acc[b] = fmaf(hs[b * DMLP + k], w, acc[b]);
  }
  float bn = bb2[n];
#pragma unroll
  for (int b = 0; b < B_; b++) biasOut[(size_t)b * LL + n] = acc[b] + bn;
}

// ---------------------------------------------------------------------------
// Fused scores -> mask(1e-9) -> softmax -> +bias -> attn(bf16).
// Block: one batch, 16 query rows, all 512 cols (33 KB LDS strip).
// A fragments (q) preloaded once into registers, reused over 4 N tiles.
// ---------------------------------------------------------------------------
__global__ __launch_bounds__(256) void scores_softmax(
    const unsigned short* __restrict__ qb, const unsigned short* __restrict__ kb,
    const unsigned char* __restrict__ mask, const float* __restrict__ biasBuf,
    unsigned short* __restrict__ attn) {
  __shared__ float s[16][L_ + 4];
  __shared__ float red[16][16];

  const int bb = blockIdx.y;
  const int l0 = blockIdx.x * 16;
  const int tid = threadIdx.x;
  const int lane = tid & 31, wave = tid >> 5;
  const int lm = lane & 15, kg = lane >> 4;

  // Preload the 4 K-step A fragments of q once (reused across all 4 N tiles).
  const unsigned short* qrow = qb + (size_t)(bb * L_ + l0 + lm) * DQ;
  V16U afrag[4];
#pragma unroll
  for (int ki = 0; ki < 4; ki++) {
    int k0 = ki * 32;
    afrag[ki].h[0] = *(const v8bf*)(qrow + k0 + kg * 16);
    afrag[ki].h[1] = *(const v8bf*)(qrow + k0 + 8 + kg * 16);
  }

#pragma unroll
  for (int j = 0; j < 4; j++) {
    const int n0 = (wave + 8 * j) * 16;
    const unsigned short* krow = kb + (size_t)(bb * L_ + n0 + lm) * DQ;  // B = k^T (row-major k)
    v8f acc = {};
#pragma unroll
    for (int ki = 0; ki < 4; ki++) {
      v16bf b = *(const v16bf*)(krow + ki * 32 + 16 * kg);
      acc = __builtin_amdgcn_wmma_f32_16x16x32_bf16(false, afrag[ki].v, false, b,
                                                    (short)0, acc, false, false);
    }
#pragma unroll
    for (int e = 0; e < 8; e++) s[e + 8 * kg][n0 + lm] = acc[e];
  }
  __syncthreads();

  // 16 threads per row, 32 cols each
  const int r = tid >> 4, c = tid & 15;
  const int col0 = c * 32;
  const unsigned char* mrow = mask + (size_t)bb * LL + (size_t)(l0 + r) * L_;

  float lmax = -3.4e38f;
  for (int j = 0; j < 32; j++) {
    int col = col0 + j;
    float v = s[r][col];
    if (mrow[col]) v = 1e-9f;    // masked_fill with 1e-9 (NOT -inf), per reference
    s[r][col] = v;
    lmax = fmaxf(lmax, v);
  }
  red[r][c] = lmax;
  __syncthreads();
  float rm = red[r][0];
#pragma unroll
  for (int j = 1; j < 16; j++) rm = fmaxf(rm, red[r][j]);

  float lsum = 0.0f;
  for (int j = 0; j < 32; j++) {
    int col = col0 + j;
    float e = __expf(s[r][col] - rm);
    s[r][col] = e;
    lsum += e;
  }
  __syncthreads();
  red[r][c] = lsum;
  __syncthreads();
  float rs = red[r][0];
#pragma unroll
  for (int j = 1; j < 16; j++) rs += red[r][j];
  float inv = 1.0f / rs;

  const float* brow = biasBuf + (size_t)bb * LL + (size_t)(l0 + r) * L_;
  unsigned short* arow = attn + (size_t)bb * LL + (size_t)(l0 + r) * L_;
  for (int j = 0; j < 32; j++) {
    int col = col0 + j;
    arow[col] = f2bfu(s[r][col] * inv + brow[col]);  // bias added AFTER softmax
  }
}

// ---------------------------------------------------------------------------
// out = attn @ v  per batch (M=512, N=128, K=512). attn strip staged in LDS
// (shared by all 8 waves); B operand streams from vT[d][m].
// ---------------------------------------------------------------------------
__global__ __launch_bounds__(256) void out_gemm(const unsigned short* __restrict__ attn,
                                                const unsigned short* __restrict__ vT,
                                                float* __restrict__ out) {
  __shared__ __align__(32) unsigned short as_[16][L_ + 8];  // 16x520 u16, pitch 1040B

  const int bb = blockIdx.y;
  const int l0 = blockIdx.x * 16;
  const int tid = threadIdx.x;

  // Cooperative stage of the 16x512 bf16 attn strip (uint4 = 8 bf16 per op).
  for (int i = tid; i < 16 * (L_ / 8); i += 256) {
    int r = i >> 6;          // L_/8 = 64 uint4 per row
    int c = (i & 63) * 8;
    *(uint4*)&as_[r][c] =
        *(const uint4*)(attn + (size_t)bb * LL + (size_t)(l0 + r) * L_ + c);
  }
  __syncthreads();

  const int lane = tid & 31, wave = tid >> 5;
  const int lm = lane & 15, kg = lane >> 4;
  const int n0 = wave * 16;  // d tile

  const unsigned short* brow = vT + (size_t)bb * DQ * L_ + (size_t)(n0 + lm) * L_;

  v8f acc = {};
  for (int k0 = 0; k0 < L_; k0 += 32) {
    V16U a;
    a.h[0] = *(const v8bf*)&as_[lm][k0 + kg * 16];
    a.h[1] = *(const v8bf*)&as_[lm][k0 + 8 + kg * 16];
    v16bf b = *(const v16bf*)(brow + k0 + 16 * kg);
    acc = __builtin_amdgcn_wmma_f32_16x16x32_bf16(false, a.v, false, b,
                                                  (short)0, acc, false, false);
  }
#pragma unroll
  for (int e = 0; e < 8; e++) {
    int l = l0 + e + 8 * kg;
    out[(size_t)(bb * L_ + l) * DQ + n0 + lm] = acc[e];
  }
}

// ---------------------------------------------------------------------------
extern "C" void kernel_launch(void* const* d_in, const int* in_sizes, int n_in,
                              void* d_out, int out_size, void* d_ws, size_t ws_size,
                              hipStream_t stream) {
  const float* query = (const float*)d_in[0];
  const float* key_  = (const float*)d_in[1];
  const float* value = (const float*)d_in[2];
  const float* sf    = (const float*)d_in[3];
  const unsigned char* mask = (const unsigned char*)d_in[4];  // jnp.bool_: 1 byte/elem
  const float* Wq  = (const float*)d_in[5];
  const float* bq  = (const float*)d_in[6];
  const float* Wk  = (const float*)d_in[7];
  const float* bk  = (const float*)d_in[8];
  const float* Wv  = (const float*)d_in[9];
  const float* bv  = (const float*)d_in[10];
  const float* Wb1 = (const float*)d_in[11];
  const float* bb1 = (const float*)d_in[12];
  const float* Wb2 = (const float*)d_in[13];
  const float* bb2 = (const float*)d_in[14];
  float* out = (float*)d_out;

  char* ws = (char*)d_ws;
  size_t off = 0;
  auto alloc = [&](size_t bytes) -> char* {
    char* p = ws + off;
    off = (off + bytes + 255) & ~(size_t)255;
    return p;
  };
  unsigned short* WqT  = (unsigned short*)alloc((size_t)DQ * DIN * 2);
  unsigned short* WkT  = (unsigned short*)alloc((size_t)DQ * DIN * 2);
  unsigned short* WvT  = (unsigned short*)alloc((size_t)DQ * DIN * 2);
  unsigned short* qb   = (unsigned short*)alloc((size_t)B_ * L_ * DQ * 2);
  unsigned short* kb   = (unsigned short*)alloc((size_t)B_ * L_ * DQ * 2);
  unsigned short* vT   = (unsigned short*)alloc((size_t)B_ * DQ * L_ * 2);
  float*          h    = (float*)alloc((size_t)B_ * DMLP * 4);
  float*          bias = (float*)alloc((size_t)B_ * LL * 4);
  unsigned short* attn = (unsigned short*)alloc((size_t)B_ * LL * 2);
  (void)ws_size; (void)in_sizes; (void)n_in; (void)out_size;

  const dim3 blk(256);
  const int wElems = DIN * DQ;

  pack_wT<<<dim3((wElems + 255) / 256), blk, 0, stream>>>(Wq, WqT, DIN, DQ);
  pack_wT<<<dim3((wElems + 255) / 256), blk, 0, stream>>>(Wk, WkT, DIN, DQ);
  pack_wT<<<dim3((wElems + 255) / 256), blk, 0, stream>>>(Wv, WvT, DIN, DQ);

  const int mtiles = (B_ * L_) / 16;  // 1024
  qkv_gemm<<<dim3(mtiles), blk, 0, stream>>>(query, WqT, bq, qb, nullptr, 0);
  qkv_gemm<<<dim3(mtiles), blk, 0, stream>>>(key_,  WkT, bk, kb, nullptr, 1);
  qkv_gemm<<<dim3(mtiles), blk, 0, stream>>>(value, WvT, bv, nullptr, vT, 2);

  mlp1<<<dim3((B_ * DMLP + 255) / 256), blk, 0, stream>>>(sf, Wb1, bb1, h);
  bias2<<<dim3(LL / 256), blk, 0, stream>>>(h, Wb2, bb2, bias);

  scores_softmax<<<dim3(L_ / 16, B_), blk, 0, stream>>>(qb, kb, mask, bias, attn);
  out_gemm<<<dim3(L_ / 16, B_), blk, 0, stream>>>(attn, vT, out);
}
